// ProgSGMultimodalModel_37434934952304
// MI455X (gfx1250) — compile-verified, matrix operand order checked by reading, not run on hardware
//
#include <hip/hip_runtime.h>

#define D_  256
#define H_  4
#define C_  64
#define L_  4
#define QS  1024   // fused q|k|v|s row stride
#define KK  256    // GEMM K (both GEMMs)
#define KP  264    // LDS row stride in shorts (16B pad -> conflict-free b128 reads)

#if defined(__has_builtin)
#if __has_builtin(__builtin_amdgcn_global_load_async_to_lds_b128) && \
    __has_builtin(__builtin_amdgcn_s_wait_asynccnt)
#define USE_ASYNC_LDS 1
#endif
#endif

typedef __attribute__((ext_vector_type(16))) __bf16          v16bf;
typedef __attribute__((ext_vector_type(16))) unsigned short  v16us;
typedef __attribute__((ext_vector_type(8)))  float           v8f;

// types for the async-to-LDS builtin: (v4i __device__*, v4i __shared__*, imm, imm)
typedef int v4i __attribute__((__vector_size__(4 * sizeof(int))));
typedef __attribute__((address_space(1))) v4i as1_v4i;
typedef __attribute__((address_space(3))) v4i as3_v4i;

struct __attribute__((aligned(16))) US8 { unsigned short s[8]; };

__device__ __forceinline__ unsigned short f2bf(float f) {
    unsigned u = __float_as_uint(f);
    u += 0x7FFFu + ((u >> 16) & 1u);           // round-to-nearest-even
    return (unsigned short)(u >> 16);
}
__device__ __forceinline__ unsigned encf(float f) {   // order-preserving float->uint
    unsigned u = __float_as_uint(f);
    return (u & 0x80000000u) ? ~u : (u | 0x80000000u);
}
__device__ __forceinline__ float decf(unsigned u) {
    return __uint_as_float((u & 0x80000000u) ? (u & 0x7FFFFFFFu) : ~u);
}

// ---------------- elementwise / setup kernels ----------------

__global__ void zero_u32(unsigned* p, long n) {
    long i = (long)blockIdx.x * blockDim.x + threadIdx.x;
    if (i < n) p[i] = 0u;
}

__global__ void embed_nodes(const int* __restrict__ x, const float* __restrict__ emb,
                            float* __restrict__ h) {
    int n = blockIdx.x, d = threadIdx.x;
    float s = 0.f;
#pragma unroll
    for (int f = 0; f < 3; ++f)
        s += emb[((size_t)f * 64 + x[n * 3 + f]) * D_ + d];
    h[(size_t)n * D_ + d] = s;
}

__global__ void embed_edges(const int* __restrict__ ea, const float* __restrict__ emb,
                            float* __restrict__ e) {
    int n = blockIdx.x, d = threadIdx.x;
    float s = 0.f;
#pragma unroll
    for (int f = 0; f < 2; ++f)
        s += emb[((size_t)f * 16 + ea[n * 2 + f]) * D_ + d];
    e[(size_t)n * D_ + d] = s;
}

__global__ void to_bf16(const float* __restrict__ a, unsigned short* __restrict__ b, long n) {
    long i = (long)blockIdx.x * blockDim.x + threadIdx.x;
    if (i < n) b[i] = f2bf(a[i]);
}

// single weight transpose: Wt[n*K + k] = bf16( W[k*Nout + n] )
__global__ void w_transpose_bf16(const float* __restrict__ W, unsigned short* __restrict__ Wt,
                                 int K, int Nout) {
    int idx = blockIdx.x * blockDim.x + threadIdx.x;
    if (idx >= K * Nout) return;
    int n = idx / K, k = idx % K;
    Wt[idx] = f2bf(W[(size_t)k * Nout + n]);
}

// fused q|k|v|s transposed weight: Wt[j*256 + k] = bf16( Wsel[k*256 + (j&255)] ), j in [0,1024)
__global__ void wcat_transpose_bf16(const float* __restrict__ Wq, const float* __restrict__ Wk,
                                    const float* __restrict__ Wv, const float* __restrict__ Wsp,
                                    unsigned short* __restrict__ Wt) {
    int idx = blockIdx.x * blockDim.x + threadIdx.x;   // over 1024*256
    if (idx >= QS * D_) return;
    int j = idx >> 8, k = idx & 255;
    int sel = j >> 8;
    const float* W = (sel == 0) ? Wq : (sel == 1) ? Wk : (sel == 2) ? Wv : Wsp;
    Wt[idx] = f2bf(W[(size_t)k * D_ + (j & 255)]);
}

__global__ void bcat_copy(const float* __restrict__ bq, const float* __restrict__ bk,
                          const float* __restrict__ bv, const float* __restrict__ bs,
                          float* __restrict__ bcat) {
    int j = blockIdx.x * blockDim.x + threadIdx.x;
    if (j >= QS) return;
    int sel = j >> 8;
    const float* b = (sel == 0) ? bq : (sel == 1) ? bk : (sel == 2) ? bv : bs;
    bcat[j] = b[j & 255];
}

// ---------------- WMMA bf16 GEMM with LDS-staged weights ----------------
// C[M,Nout] = A[M,256] * B[256,Nout] + bias.  A: bf16 [M,256]; Bt: bf16 [Nout,256].
// Block = 8 waves: one 64-column group, 8 consecutive 16-row tiles.
// Weight strip (64 x 256 bf16) staged in LDS (async path when available).

__device__ __forceinline__ v16bf load_frag(const unsigned short* __restrict__ row,
                                           int k0, int half) {
    US8 lo = *reinterpret_cast<const US8*>(row + k0 + 8 * half);        // K = k0+8h .. +7
    US8 hi = *reinterpret_cast<const US8*>(row + k0 + 16 + 8 * half);   // K = k0+16+8h .. +7
    v16us t;
#pragma unroll
    for (int i = 0; i < 8; ++i) { t[i] = lo.s[i]; t[8 + i] = hi.s[i]; }
    return __builtin_bit_cast(v16bf, t);
}

__device__ __forceinline__ v16bf load_frag_lds(const unsigned short* row, int k0, int half) {
    US8 lo = *reinterpret_cast<const US8*>(row + k0 + 8 * half);
    US8 hi = *reinterpret_cast<const US8*>(row + k0 + 16 + 8 * half);
    v16us t;
#pragma unroll
    for (int i = 0; i < 8; ++i) { t[i] = lo.s[i]; t[8 + i] = hi.s[i]; }
    return __builtin_bit_cast(v16bf, t);
}

__device__ __forceinline__ void store_tile(v8f acc, float* __restrict__ Cmat,
                                           const float* __restrict__ bias,
                                           int m0, int n, int M, int Nout, int half) {
    float bv = bias ? bias[n] : 0.f;
#pragma unroll
    for (int v = 0; v < 8; ++v) {
        int m = m0 + v + 8 * half;
        if (m < M) Cmat[(size_t)m * Nout + n] = acc[v] + bv;
    }
}

__global__ void wmma_gemm_bf16_lds(const unsigned short* __restrict__ A,
                                   const unsigned short* __restrict__ Bt,
                                   const float* __restrict__ bias,
                                   float* __restrict__ Cmat,
                                   int M, int Nout) {
    __shared__ unsigned short bsh[64 * KP];            // 33 KB

    int tilesNg = Nout >> 6;
    int mblk = blockIdx.x / tilesNg;
    int tg   = blockIdx.x % tilesNg;

    // ---- cooperative stage of the 64x256 weight strip into LDS ----
    const unsigned short* bsrc = Bt + (size_t)tg * 64 * KK;
#ifdef USE_ASYNC_LDS
    {
        const char* gsrc = (const char*)bsrc;
        char* lbase = (char*)bsh;
#pragma unroll
        for (int i = 0; i < 8; ++i) {
            int c = threadIdx.x + i * 256;             // 16B chunk id, 2048 total
            int row = c >> 5, col = c & 31;
            unsigned goff = (unsigned)(row * KK + col * 8) * 2;
            unsigned loff = (unsigned)(row * KP + col * 8) * 2;
            __builtin_amdgcn_global_load_async_to_lds_b128(
                (as1_v4i*)(gsrc + goff), (as3_v4i*)(lbase + loff), 0, 0);
        }
        __builtin_amdgcn_s_wait_asynccnt(0);
    }
#else
#pragma unroll
    for (int i = 0; i < 8; ++i) {
        int c = threadIdx.x + i * 256;
        int row = c >> 5, col = c & 31;
        *reinterpret_cast<US8*>(bsh + row * KP + col * 8) =
            *reinterpret_cast<const US8*>(bsrc + row * KK + col * 8);
    }
#endif
    __syncthreads();

    int wv   = threadIdx.x >> 5;
    int lane = threadIdx.x & 31;
    int half = lane >> 4;
    int lrow = lane & 15;
    int tm   = mblk * 8 + wv;
    bool active = (tm * 16 < M);

    int ar = tm * 16 + lrow; if (ar >= M) ar = M - 1;
    const unsigned short* arow = A + (size_t)ar * KK;
    const unsigned short* l0 = bsh + (     lrow) * KP;
    const unsigned short* l1 = bsh + (16 + lrow) * KP;
    const unsigned short* l2 = bsh + (32 + lrow) * KP;
    const unsigned short* l3 = bsh + (48 + lrow) * KP;

    v8f acc0 = {}, acc1 = {}, acc2 = {}, acc3 = {};
    for (int k0 = 0; k0 < KK; k0 += 32) {
        v16bf a  = load_frag(arow, k0, half);
        v16bf f0 = load_frag_lds(l0, k0, half);
        v16bf f1 = load_frag_lds(l1, k0, half);
        v16bf f2 = load_frag_lds(l2, k0, half);
        v16bf f3 = load_frag_lds(l3, k0, half);
        acc0 = __builtin_amdgcn_wmma_f32_16x16x32_bf16(false, a, false, f0, (short)0, acc0, false, false);
        acc1 = __builtin_amdgcn_wmma_f32_16x16x32_bf16(false, a, false, f1, (short)0, acc1, false, false);
        acc2 = __builtin_amdgcn_wmma_f32_16x16x32_bf16(false, a, false, f2, (short)0, acc2, false, false);
        acc3 = __builtin_amdgcn_wmma_f32_16x16x32_bf16(false, a, false, f3, (short)0, acc3, false, false);
    }

    if (active) {
        int m0 = tm * 16;
        store_tile(acc0, Cmat, bias, m0, tg * 64 +      lrow, M, Nout, half);
        store_tile(acc1, Cmat, bias, m0, tg * 64 + 16 + lrow, M, Nout, half);
        store_tile(acc2, Cmat, bias, m0, tg * 64 + 32 + lrow, M, Nout, half);
        store_tile(acc3, Cmat, bias, m0, tg * 64 + 48 + lrow, M, Nout, half);
    }
}

// ---------------- attention (segment softmax over dst) ----------------
// qkvs rows: [ q(0..255) | k(256..511) | v(512..767) | s(768..1023) ], stride QS

__global__ void edge_logits(const int* __restrict__ ei, const float* __restrict__ qkvs,
                            const float* __restrict__ ee,
                            float* __restrict__ logits, unsigned* __restrict__ menc, int E) {
    int idx = blockIdx.x * blockDim.x + threadIdx.x;
    if (idx >= E * H_) return;
    int e = idx >> 2, hd = idx & 3;
    int src = ei[e], dst = ei[E + e];
    const float4* qp = (const float4*)(qkvs + (size_t)dst * QS +        hd * C_);
    const float4* kp = (const float4*)(qkvs + (size_t)src * QS + D_  +  hd * C_);
    const float4* ep = (const float4*)(ee   + (size_t)e   * D_ +        hd * C_);
    float acc = 0.f;
#pragma unroll
    for (int i = 0; i < 16; ++i) {
        float4 a = qp[i], b = kp[i], c = ep[i];
        acc += a.x*(b.x+c.x) + a.y*(b.y+c.y) + a.z*(b.z+c.z) + a.w*(b.w+c.w);
    }
    float lg = acc * 0.125f;                       // 1/sqrt(64)
    logits[idx] = lg;
    atomicMax(&menc[dst * H_ + hd], encf(lg));
}

__global__ void finalize_max(const unsigned* __restrict__ menc, float* __restrict__ m, long n) {
    long i = (long)blockIdx.x * blockDim.x + threadIdx.x;
    if (i < n) { unsigned u = menc[i]; m[i] = (u == 0u) ? 0.f : decf(u); }
}

__global__ void edge_exp(const int* __restrict__ ei, const float* __restrict__ m,
                         float* __restrict__ logits, float* __restrict__ denom, int E) {
    int idx = blockIdx.x * blockDim.x + threadIdx.x;
    if (idx >= E * H_) return;
    int e = idx >> 2, hd = idx & 3;
    int dst = ei[E + e];
    float p = __expf(logits[idx] - m[dst * H_ + hd]);
    logits[idx] = p;
    atomicAdd(&denom[dst * H_ + hd], p);
}

__global__ void edge_scatter(const int* __restrict__ ei, const float* __restrict__ p,
                             const float* __restrict__ qkvs, const float* __restrict__ ee,
                             float* __restrict__ aout, int E) {
    long idx = (long)blockIdx.x * blockDim.x + threadIdx.x;
    if (idx >= (long)E * D_) return;
    int e = (int)(idx >> 8), c = (int)(idx & 255), hd = c >> 6;
    int src = ei[e], dst = ei[E + e];
    float v = qkvs[(size_t)src * QS + 2 * D_ + c] + ee[(size_t)e * D_ + c];
    atomicAdd(&aout[(size_t)dst * D_ + c], p[e * H_ + hd] * v);
}

// ---------------- per-node: normalize, beta gate, relu, LN, residual ----------------

__device__ __forceinline__ float block_reduce(float v, float* red) {
    int tid = threadIdx.x;
    red[tid] = v; __syncthreads();
    for (int s = 128; s > 0; s >>= 1) {
        if (tid < s) red[tid] += red[tid + s];
        __syncthreads();
    }
    float r = red[0]; __syncthreads();
    return r;
}

__global__ void node_update(float* __restrict__ h, const float* __restrict__ aout,
                            const float* __restrict__ denom, const float* __restrict__ qkvs,
                            const float* __restrict__ Wb, const float* __restrict__ g,
                            const float* __restrict__ b) {
    __shared__ float red[256];
    int n = blockIdx.x, c = threadIdx.x;
    float o  = aout[(size_t)n * D_ + c] / (denom[n * H_ + (c >> 6)] + 1e-16f);
    float xv = qkvs[(size_t)n * QS + 3 * D_ + c];
    float part = o * Wb[c] + xv * Wb[D_ + c] + (o - xv) * Wb[2 * D_ + c];
    float s = block_reduce(part, red);
    float beta = 1.f / (1.f + __expf(-s));
    float hn = beta * xv + (1.f - beta) * o;
    hn = fmaxf(hn, 0.f);
    float mean = block_reduce(hn, red) * (1.f / D_);
    float d = hn - mean;
    float var = block_reduce(d * d, red) * (1.f / D_);
    float y = d * __frsqrt_rn(var + 1e-5f) * g[c] + b[c];
    h[(size_t)n * D_ + c] += y;
}

// ---------------- pooling + head MLPs ----------------

__global__ void pool_sum(const float* __restrict__ h, const int* __restrict__ batch,
                         float* __restrict__ pooled, int N) {
    long idx = (long)blockIdx.x * blockDim.x + threadIdx.x;
    if (idx >= (long)N * D_) return;
    int n = (int)(idx >> 8), c = (int)(idx & 255);
    atomicAdd(&pooled[(size_t)batch[n] * D_ + c], h[idx]);
}

__global__ void head_mlp(const float* __restrict__ pooled, const float* __restrict__ scalars,
                         const float* __restrict__ codep,
                         const float* Wp1, const float* bp1, const float* Wp2, const float* bp2,
                         const float* Wc,  const float* bc,
                         const float* Wf1, const float* bf1, const float* Wf2, const float* bf2,
                         const float* Wf3, const float* bf3, float* __restrict__ out) {
    __shared__ float fused[2 * D_];
    __shared__ float buf[D_];
    __shared__ float z2[D_ / 2];
    int g = blockIdx.x, d = threadIdx.x;

    float acc = bp1[d];
#pragma unroll
    for (int k = 0; k < 5; ++k) acc += scalars[g * 5 + k] * Wp1[k * D_ + d];
    buf[d] = fmaxf(acc, 0.f);
    __syncthreads();

    float a2 = bp2[d];
    for (int k = 0; k < D_; ++k) a2 += buf[k] * Wp2[k * D_ + d];
    float pc = pooled[(size_t)g * D_ + d] + a2;

    float a3 = bc[d];
    for (int k = 0; k < 768; ++k) a3 += codep[(size_t)g * 768 + k] * Wc[k * D_ + d];
    __syncthreads();
    fused[d] = pc; fused[D_ + d] = a3;
    __syncthreads();

    float a4 = bf1[d];
    for (int k = 0; k < 2 * D_; ++k) a4 += fused[k] * Wf1[k * D_ + d];
    buf[d] = fmaxf(a4, 0.f);
    __syncthreads();

    if (d < D_ / 2) {
        float a5 = bf2[d];
        for (int k = 0; k < D_; ++k) a5 += buf[k] * Wf2[k * (D_ / 2) + d];
        z2[d] = fmaxf(a5, 0.f);
    }
    __syncthreads();
    if (d < 4) {
        float a6 = bf3[d];
        for (int k = 0; k < D_ / 2; ++k) a6 += z2[k] * Wf3[k * 4 + d];
        out[g * 4 + d] = a6;
    }
}

// ---------------- launcher ----------------

extern "C" void kernel_launch(void* const* d_in, const int* in_sizes, int n_in,
                              void* d_out, int out_size, void* d_ws, size_t ws_size,
                              hipStream_t stream) {
    const int N = in_sizes[0] / 3;
    const int E = in_sizes[1] / 2;
    const int G = in_sizes[4] / 5;

    const int*   x        = (const int*)  d_in[0];
    const int*   ei       = (const int*)  d_in[1];
    const int*   ea       = (const int*)  d_in[2];
    const int*   batch    = (const int*)  d_in[3];
    const float* scalars  = (const float*)d_in[4];
    const float* codep    = (const float*)d_in[5];
    const float* node_emb = (const float*)d_in[7];
    const float* edge_emb = (const float*)d_in[8];
    const float* Wq = (const float*)d_in[9],  *bq = (const float*)d_in[10];
    const float* Wk = (const float*)d_in[11], *bk = (const float*)d_in[12];
    const float* Wv = (const float*)d_in[13], *bv = (const float*)d_in[14];
    const float* We = (const float*)d_in[15], *be = (const float*)d_in[16];
    const float* Ws = (const float*)d_in[17], *bs = (const float*)d_in[18];
    const float* Wbeta = (const float*)d_in[19];
    const float* lng = (const float*)d_in[20], *lnb = (const float*)d_in[21];
    const float* Wp1 = (const float*)d_in[22], *bp1 = (const float*)d_in[23];
    const float* Wp2 = (const float*)d_in[24], *bp2 = (const float*)d_in[25];
    const float* Wc  = (const float*)d_in[26], *bc  = (const float*)d_in[27];
    const float* Wf1 = (const float*)d_in[28], *bf1 = (const float*)d_in[29];
    const float* Wf2 = (const float*)d_in[30], *bf2 = (const float*)d_in[31];
    const float* Wf3 = (const float*)d_in[32], *bf3 = (const float*)d_in[33];
    float* out = (float*)d_out;

    // workspace carve-up (256B aligned)
    size_t off = 0;
    auto alloc = [&](size_t bytes) -> void* {
        void* p = (char*)d_ws + off;
        off += (bytes + 255) & ~(size_t)255;
        return p;
    };
    float* h    = (float*)alloc((size_t)N * D_ * 4);
    float* e0   = (float*)alloc((size_t)E * D_ * 4);
    unsigned short* hb  = (unsigned short*)alloc((size_t)N * D_ * 2);
    unsigned short* eb  = (unsigned short*)alloc((size_t)E * D_ * 2);
    unsigned short* WtQ = (unsigned short*)alloc((size_t)QS * D_ * 2);  // fused qkvs weights^T
    unsigned short* WtE = (unsigned short*)alloc((size_t)D_ * D_ * 2);  // edge weights^T
    float* bcat = (float*)alloc((size_t)QS * 4);
    float* qkvs = (float*)alloc((size_t)N * QS * 4);
    float* ee   = (float*)alloc((size_t)E * D_ * 4);
    float* lg   = (float*)alloc((size_t)E * H_ * 4);
    unsigned* menc = (unsigned*)alloc((size_t)N * H_ * 4);
    float* mmax = (float*)alloc((size_t)N * H_ * 4);
    float* den  = (float*)alloc((size_t)N * H_ * 4);
    float* aout = (float*)alloc((size_t)N * D_ * 4);
    float* pooled = (float*)alloc((size_t)G * D_ * 4);

    const int T = 256;
    auto blks = [](long n, int t) { return (unsigned)((n + t - 1) / t); };

    embed_nodes<<<N, T, 0, stream>>>(x, node_emb, h);
    embed_edges<<<E, T, 0, stream>>>(ea, edge_emb, e0);
    to_bf16<<<blks((long)E * D_, T), T, 0, stream>>>(e0, eb, (long)E * D_);

    // block = 8 waves = 128 rows x 64 cols
    const unsigned gemmBlkQ = (unsigned)(((N + 127) / 128) * (QS / 64));
    const unsigned gemmBlkE = (unsigned)(((E + 127) / 128) * (D_ / 64));

    for (int l = 0; l < L_; ++l) {
        const size_t wo = (size_t)l * D_ * D_;
        const size_t bo = (size_t)l * D_;

        to_bf16<<<blks((long)N * D_, T), T, 0, stream>>>(h, hb, (long)N * D_);

        wcat_transpose_bf16<<<blks((long)QS * D_, T), T, 0, stream>>>(
            Wq + wo, Wk + wo, Wv + wo, Ws + wo, WtQ);
        bcat_copy<<<blks(QS, T), T, 0, stream>>>(bq + bo, bk + bo, bv + bo, bs + bo, bcat);
        wmma_gemm_bf16_lds<<<gemmBlkQ, T, 0, stream>>>(hb, WtQ, bcat, qkvs, N, QS);

        w_transpose_bf16<<<blks((long)D_ * D_, T), T, 0, stream>>>(We + wo, WtE, D_, D_);
        wmma_gemm_bf16_lds<<<gemmBlkE, T, 0, stream>>>(eb, WtE, be + bo, ee, E, D_);

        zero_u32<<<blks((long)N * H_, T), T, 0, stream>>>(menc, (long)N * H_);
        zero_u32<<<blks((long)N * H_, T), T, 0, stream>>>((unsigned*)den, (long)N * H_);
        zero_u32<<<blks((long)N * D_, T), T, 0, stream>>>((unsigned*)aout, (long)N * D_);

        edge_logits<<<blks((long)E * H_, T), T, 0, stream>>>(ei, qkvs, ee, lg, menc, E);
        finalize_max<<<blks((long)N * H_, T), T, 0, stream>>>(menc, mmax, (long)N * H_);
        edge_exp<<<blks((long)E * H_, T), T, 0, stream>>>(ei, mmax, lg, den, E);
        edge_scatter<<<blks((long)E * D_, T), T, 0, stream>>>(ei, lg, qkvs, ee, aout, E);

        node_update<<<N, T, 0, stream>>>(h, aout, den, qkvs,
                                         Wbeta + (size_t)l * 3 * D_, lng + bo, lnb + bo);
    }

    zero_u32<<<blks((long)G * D_, T), T, 0, stream>>>((unsigned*)pooled, (long)G * D_);
    pool_sum<<<blks((long)N * D_, T), T, 0, stream>>>(h, batch, pooled, N);
    head_mlp<<<G, T, 0, stream>>>(pooled, scalars, codep, Wp1, bp1, Wp2, bp2,
                                  Wc, bc, Wf1, bf1, Wf2, bf2, Wf3, bf3, out);

    (void)n_in; (void)out_size; (void)ws_size; (void)in_sizes;
}
